// SwiGLUMoE_5712306503962
// MI455X (gfx1250) — matrix operand order repaired and checked by reading, not compiled, not run on hardware
//
#include <hip/hip_runtime.h>
#include <hip/hip_bf16.h>
#include <cstdint>

// Problem constants (fixed by the reference)
#define TN  4096   // tokens
#define TD  1024   // model dim
#define TH2 4096   // expert hidden*2
#define TH  2048   // hidden (output columns)
#define TE  8      // experts

typedef __attribute__((ext_vector_type(16))) __bf16         v16bf;
typedef __attribute__((ext_vector_type(8)))  float          v8f;
typedef __attribute__((ext_vector_type(8)))  unsigned short v8us;
typedef __attribute__((ext_vector_type(4)))  unsigned int   u32x4;
typedef __attribute__((ext_vector_type(8)))  int            i32x8;
typedef __attribute__((ext_vector_type(4)))  int            i32x4;

static __device__ __forceinline__ unsigned short f2bf(float f) {
  unsigned u = __float_as_uint(f);
  u += 0x7FFFu + ((u >> 16) & 1u);   // round-to-nearest-even
  return (unsigned short)(u >> 16);
}
static __device__ __forceinline__ float bf2f(unsigned short h) {
  return __uint_as_float(((unsigned)h) << 16);
}
static __device__ __forceinline__ v16bf ld_frag(const unsigned short* p, int second_off) {
  union { v16bf v; v8us u[2]; } f;
  f.u[0] = *(const v8us*)(p);
  f.u[1] = *(const v8us*)(p + second_off);
  return f.v;
}
// LDS byte offset of a __shared__ object (generic -> addrspace(3) -> int)
static __device__ __forceinline__ unsigned lds_off(void* p) {
  return (unsigned)(unsigned long long)(__attribute__((address_space(3))) void*)p;
}

// TDM: async-load a 2D bf16 tile (tileRows x 32 elems, row stride TD elems) into LDS
// with 16B padding after each 64B row => LDS pitch 80B (bank-conflict-free).
static __device__ __forceinline__ void tdm_load_tile(const unsigned short* gsrc,
                                                     unsigned ldsByteOff, int tileRows) {
  unsigned long long ga = (unsigned long long)gsrc;   // byte address of tile origin
  u32x4 g0;
  g0[0] = 1u;                                         // count=1, user mode, no gather
  g0[1] = ldsByteOff;                                 // lds_addr
  g0[2] = (unsigned)ga;                               // global_addr[31:0]
  g0[3] = (unsigned)((ga >> 32) & 0x01FFFFFFu) | (2u << 30);  // addr[56:32] | type=2
  i32x8 g1;
  g1[0] = (1 << 16)      // data_size = 1 (2 bytes)
        | (1 << 20)      // pad_enable
        | (3 << 22)      // pad_interval: 64B
        | (3 << 25);     // pad_amount: 4 DWORDs = 16B
  g1[1] = (int)(((unsigned)TD & 0xFFFFu) << 16);                       // tensor_dim0 lo16
  g1[2] = (int)(((unsigned)TD >> 16) & 0xFFFFu)
        | (int)((((unsigned)(TE * TH2)) & 0xFFFFu) << 16);             // dim0 hi | dim1 lo
  g1[3] = (int)((((unsigned)(TE * TH2)) >> 16) & 0xFFFFu) | (32 << 16);// dim1 hi | tile_dim0=32
  g1[4] = tileRows & 0xFFFF;                                           // tile_dim1 | tile_dim2=0
  g1[5] = TD;                                                          // tensor_dim0_stride lo32
  g1[6] = 0;
  g1[7] = 0;
  i32x4 gz = {0, 0, 0, 0};
#if __clang_major__ >= 23
  i32x8 gz8 = {0, 0, 0, 0, 0, 0, 0, 0};
  __builtin_amdgcn_tensor_load_to_lds(g0, g1, gz, gz, gz8, 0);
#else
  __builtin_amdgcn_tensor_load_to_lds(g0, g1, gz, gz, 0);
#endif
}

// ---------------- routing ----------------
__global__ void moe_init(int* counts, int* cursors) {
  int t = threadIdx.x;
  if (t < TE) { counts[t] = 0; cursors[t] = 0; }
}
__global__ void moe_count(const int* __restrict__ idx, int* __restrict__ counts) {
  int n = blockIdx.x * 256 + threadIdx.x;
  if (n < TN) atomicAdd(&counts[idx[n]], 1);
}
__global__ void moe_scan(const int* __restrict__ counts, int* __restrict__ offsets) {
  if (threadIdx.x == 0) {
    int s = 0;
    for (int e = 0; e < TE; ++e) { offsets[e] = s; s += counts[e]; }
  }
}
__global__ void moe_scatter(const int* __restrict__ idx, const int* __restrict__ offsets,
                            int* __restrict__ cursors, int* __restrict__ perm) {
  int n = blockIdx.x * 256 + threadIdx.x;
  if (n < TN) {
    int e = idx[n];
    int p = atomicAdd(&cursors[e], 1);
    perm[offsets[e] + p] = n;
  }
}

// ---------------- one-time precision split: x -> bf16 hi/lo (row-major) ----------------
__global__ __launch_bounds__(256)
void conv_x(const float* __restrict__ x, unsigned short* __restrict__ xh,
            unsigned short* __restrict__ xl) {
  size_t i = ((size_t)blockIdx.x * 256 + threadIdx.x) * 4;
  float4 v = *(const float4*)(x + i);
  float  a[4] = {v.x, v.y, v.z, v.w};
  ushort4 h, l;
  unsigned short* hp = &h.x;
  unsigned short* lp = &l.x;
  #pragma unroll
  for (int j = 0; j < 4; ++j) {
    unsigned short hh = f2bf(a[j]);
    hp[j] = hh;
    lp[j] = f2bf(a[j] - bf2f(hh));
  }
  *(ushort4*)(xh + i) = h;
  *(ushort4*)(xl + i) = l;
}

// ---------------- one-time split + transpose: W -> Wt[e][n][k] hi/lo ----------------
__global__ __launch_bounds__(256)
void conv_w(const float* __restrict__ w, unsigned short* __restrict__ wh,
            unsigned short* __restrict__ wl) {
  __shared__ unsigned short th[64][72];
  __shared__ unsigned short tl[64][72];
  const int k0 = blockIdx.x * 64;
  const int n0 = blockIdx.y * 64;
  const int e  = blockIdx.z;
  const int tid = threadIdx.x;

  #pragma unroll
  for (int it = 0; it < 4; ++it) {
    int id = tid + it * 256;
    int kk = id >> 4;
    int n4 = (id & 15) * 4;
    float4 v = *(const float4*)(w + ((size_t)(e * TD + k0 + kk)) * TH2 + n0 + n4);
    float  a[4] = {v.x, v.y, v.z, v.w};
    #pragma unroll
    for (int j = 0; j < 4; ++j) {
      unsigned short hh = f2bf(a[j]);
      th[n4 + j][kk] = hh;
      tl[n4 + j][kk] = f2bf(a[j] - bf2f(hh));
    }
  }
  __syncthreads();

  #pragma unroll
  for (int it = 0; it < 2; ++it) {
    int id = tid + it * 256;
    int nn = id >> 3;
    int kc = (id & 7) * 8;
    size_t dst = ((size_t)(e * TH2 + n0 + nn)) * TD + k0 + kc;
    *(v8us*)(wh + dst) = *(const v8us*)&th[nn][kc];
    *(v8us*)(wl + dst) = *(const v8us*)&tl[nn][kc];
  }
}

// ---------------- grouped GEMM + fused SwiGLU, TDM double-buffered ----------------
__global__ __launch_bounds__(256)
void moe_swiglu_gemm(const unsigned short* __restrict__ xh,
                     const unsigned short* __restrict__ xl,
                     const unsigned short* __restrict__ wh,  // Wt[e][n][k]
                     const unsigned short* __restrict__ wl,
                     const int*   __restrict__ perm,
                     const int*   __restrict__ counts,
                     const int*   __restrict__ offsets,
                     float*       __restrict__ out)
{
  // pitch 40 ushorts = 80B = 20 banks; matches TDM pad (64B row + 16B pad)
  __shared__ unsigned short sAh[2][64][40];
  __shared__ unsigned short sAl[2][64][40];
  __shared__ unsigned short sBh[2][256][40];   // column-major: [col][k]
  __shared__ unsigned short sBl[2][256][40];

  const int e     = blockIdx.z;
  const int cnt   = counts[e];
  const int tile0 = blockIdx.y * 64;
  if (tile0 >= cnt) return;                    // uniform exit
  const int off     = offsets[e];
  const int colTile = blockIdx.x * 128;        // in H units

  const int tid   = threadIdx.x;
  const int lane  = tid & 31;
  const int wid   = tid >> 5;
  const int waveM = wid & 1;
  const int waveN = wid >> 1;
  const int lh    = lane >> 4;
  const int li    = lane & 15;

  // A staging coords: one 16B chunk per array per thread
  const int  arow   = tid >> 2;
  const int  akc    = (tid & 3) * 8;
  const bool avalid = (tile0 + arow) < cnt;
  const int  atok   = avalid ? perm[off + tile0 + arow] : 0;
  const unsigned short* xrh = xh + (size_t)atok * TD;
  const unsigned short* xrl = xl + (size_t)atok * TD;
  const v8us uz = {0,0,0,0,0,0,0,0};

  // TDM source/dest bases for B (two contiguous 128-row groups of Wt per buffer)
  const size_t rowA0 = (size_t)(e * TH2 + colTile) * TD;        // a-half rows
  const size_t rowB0 = (size_t)(e * TH2 + TH + colTile) * TD;   // b-half rows
  const unsigned offBh = lds_off((void*)&sBh[0][0][0]);
  const unsigned offBl = lds_off((void*)&sBl[0][0][0]);
  const unsigned BUFB  = 256u * 80u;      // bytes per B buffer
  const unsigned HALFB = 128u * 80u;      // bytes per 128-row half

  v8f acc[2][2][2];
  {
    const v8f vz = {0.f,0.f,0.f,0.f,0.f,0.f,0.f,0.f};
    for (int a = 0; a < 2; ++a)
      for (int b = 0; b < 2; ++b)
        for (int c = 0; c < 2; ++c) acc[a][b][c] = vz;
  }

  // --- pipeline prologue: stage k-slice 0 into buffer 0 ---
  {
    v8us ah = avalid ? *(const v8us*)(xrh + akc) : uz;
    v8us al = avalid ? *(const v8us*)(xrl + akc) : uz;
    *(v8us*)&sAh[0][arow][akc] = ah;
    *(v8us*)&sAl[0][arow][akc] = al;
  }
  if (wid == 0) {
    tdm_load_tile(wh + rowA0, offBh + 0,     128);
    tdm_load_tile(wh + rowB0, offBh + HALFB, 128);
    tdm_load_tile(wl + rowA0, offBl + 0,     128);
    tdm_load_tile(wl + rowB0, offBl + HALFB, 128);
  }

  const int nT = TD / 32;
  for (int i = 0; i < nT; ++i) {
    const int buf = i & 1;
    const int nxt = buf ^ 1;
    if (i + 1 < nT) {
      const int kn = (i + 1) * 32;
      v8us ah = avalid ? *(const v8us*)(xrh + kn + akc) : uz;
      v8us al = avalid ? *(const v8us*)(xrl + kn + akc) : uz;
      *(v8us*)&sAh[nxt][arow][akc] = ah;
      *(v8us*)&sAl[nxt][arow][akc] = al;
      if (wid == 0) {
        tdm_load_tile(wh + rowA0 + kn, offBh + nxt * BUFB,         128);
        tdm_load_tile(wh + rowB0 + kn, offBh + nxt * BUFB + HALFB, 128);
        tdm_load_tile(wl + rowA0 + kn, offBl + nxt * BUFB,         128);
        tdm_load_tile(wl + rowB0 + kn, offBl + nxt * BUFB + HALFB, 128);
        __builtin_amdgcn_s_wait_tensorcnt(4);   // retire current buffer's 4 ops
      }
    } else if (wid == 0) {
      __builtin_amdgcn_s_wait_tensorcnt(0);     // drain for final buffer
    }
    __syncthreads();

    // ---- fragments + WMMA on `buf` ----
    v16bf aH[2], aL[2];
    #pragma unroll
    for (int ms = 0; ms < 2; ++ms) {
      int r  = waveM * 32 + ms * 16 + li;
      int kb = lh * 8;
      aH[ms] = ld_frag(&sAh[buf][r][kb], 16);
      aL[ms] = ld_frag(&sAl[buf][r][kb], 16);
    }
    #pragma unroll
    for (int half = 0; half < 2; ++half) {
      #pragma unroll
      for (int ns = 0; ns < 2; ++ns) {
        int c  = half * 128 + waveN * 32 + ns * 16 + li;
        int kb = lh * 8;
        v16bf bH = ld_frag(&sBh[buf][c][kb], 16);
        v16bf bL = ld_frag(&sBl[buf][c][kb], 16);
        #pragma unroll
        for (int ms = 0; ms < 2; ++ms) {
          v8f a0 = acc[ms][ns][half];
          a0 = __builtin_amdgcn_wmma_f32_16x16x32_bf16(false, aL[ms], false, bH, (short)0, a0, false, false);
          a0 = __builtin_amdgcn_wmma_f32_16x16x32_bf16(false, aH[ms], false, bL, (short)0, a0, false, false);
          a0 = __builtin_amdgcn_wmma_f32_16x16x32_bf16(false, aH[ms], false, bH, (short)0, a0, false, false);
          acc[ms][ns][half] = a0;
        }
      }
    }
    __syncthreads();
  }

  // ---- fused SwiGLU epilogue ----
  #pragma unroll
  for (int ms = 0; ms < 2; ++ms) {
    #pragma unroll
    for (int ns = 0; ns < 2; ++ns) {
      v8f A_ = acc[ms][ns][0];
      v8f B_ = acc[ms][ns][1];
      int col = colTile + waveN * 32 + ns * 16 + li;
      int rb  = waveM * 32 + ms * 16 + lh * 8;
      #pragma unroll
      for (int v = 0; v < 8; ++v) {
        int gr = tile0 + rb + v;
        if (gr < cnt) {
          int   tokn = perm[off + gr];
          float av   = A_[v];
          float bv   = B_[v];
          out[(size_t)tokn * TH + col] = (av / (1.0f + __expf(-av))) * bv;
        }
      }
    }
  }
}

extern "C" void kernel_launch(void* const* d_in, const int* in_sizes, int n_in,
                              void* d_out, int out_size, void* d_ws, size_t ws_size,
                              hipStream_t stream) {
  const float* x   = (const float*)d_in[0];
  const float* w   = (const float*)d_in[1];
  const int*   idx = (const int*)d_in[2];
  float*       out = (float*)d_out;

  char* base   = (char*)d_ws;
  int*  counts  = (int*)base;          // [8]
  int*  cursors = counts + 8;          // [8]
  int*  offsets = counts + 16;         // [8]
  int*  perm    = counts + 32;         // [TN]
  size_t intBytes = (size_t)(32 + TN) * sizeof(int);
  size_t o = (intBytes + 255) & ~(size_t)255;

  unsigned short* xh = (unsigned short*)(base + o);  o += (size_t)TN * TD * 2;
  unsigned short* xl = (unsigned short*)(base + o);  o += (size_t)TN * TD * 2;
  unsigned short* wh = (unsigned short*)(base + o);  o += (size_t)TE * TH2 * TD * 2;
  unsigned short* wl = (unsigned short*)(base + o);

  moe_init   <<<1, 32, 0, stream>>>(counts, cursors);
  moe_count  <<<(TN + 255) / 256, 256, 0, stream>>>(idx, counts);
  moe_scan   <<<1, 1, 0, stream>>>(counts, offsets);
  moe_scatter<<<(TN + 255) / 256, 256, 0, stream>>>(idx, offsets, cursors, perm);

  conv_x<<<(TN * TD) / 1024, 256, 0, stream>>>(x, xh, xl);
  conv_w<<<dim3(TD / 64, TH2 / 64, TE), 256, 0, stream>>>(w, wh, wl);

  dim3 grid(TH / 128, (TN + 63) / 64, TE);
  moe_swiglu_gemm<<<grid, 256, 0, stream>>>(xh, xl, wh, wl, perm, counts, offsets, out);
}
